// Attention_22325240004803
// MI455X (gfx1250) — matrix operand order repaired and checked by reading, not compiled, not run on hardware
//
#include <hip/hip_runtime.h>
#include <hip/hip_bf16.h>

// ---- problem constants ----
#define Bq   4
#define Nseq 1024
#define Cdim 768
#define Hh   12
#define HDd  64
#define M_QKV   4096            // B*N
#define N_QKV   2304            // 3*C
#define K_DIM   768             // C

typedef __attribute__((ext_vector_type(16))) __bf16 v16bf;
typedef __attribute__((ext_vector_type(8)))  __bf16 v8bf;
typedef __attribute__((ext_vector_type(8)))  float  v8f;
typedef __attribute__((ext_vector_type(4)))  int    v4i;

// ---- async global->LDS staging (CDNA5 ASYNCcnt path), with safe fallback ----
// Builtin signature (from hipcc diagnostic): (v4i AS1* src, v4i AS3* dst, imm, imm)
#define AS1V4(p) ((__attribute__((address_space(1))) v4i*)(p))
#define AS3V4(p) ((__attribute__((address_space(3))) v4i*)(p))

#if defined(__has_builtin)
#if __has_builtin(__builtin_amdgcn_global_load_async_to_lds_b128)
#define HAVE_ASYNC_LDS 1
#endif
#endif

#ifdef HAVE_ASYNC_LDS
// per-lane 16B copy: LDS[dst] = MEM[src], tracked by ASYNCcnt
#define STAGE16(dst, src) \
  __builtin_amdgcn_global_load_async_to_lds_b128(AS1V4(src), AS3V4(dst), 0, 0)
#define STAGE_FENCE() asm volatile("s_wait_asynccnt 0" ::: "memory")
#else
#define STAGE16(dst, src) (*(v8bf*)(dst) = *(const v8bf*)(src))
#define STAGE_FENCE()
#endif

// ---- bf16 helpers (manual RNE; storage as ushort) ----
__device__ __forceinline__ unsigned short f2bf(float f) {
  union { float f; unsigned u; } v; v.f = f;
  unsigned u = v.u;
  unsigned r = u >> 16;
  unsigned rem = u & 0xffffu;
  r += (rem > 0x8000u) || (rem == 0x8000u && (r & 1u));
  return (unsigned short)r;
}
__device__ __forceinline__ float bf2f(unsigned short h) {
  union { unsigned u; float f; } v; v.u = ((unsigned)h) << 16;
  return v.f;
}
__device__ __forceinline__ v16bf cat8(v8bf lo, v8bf hi) {
  return __builtin_shufflevector(lo, hi, 0,1,2,3,4,5,6,7,8,9,10,11,12,13,14,15);
}
__device__ __forceinline__ v8f wmma_bf16(v16bf a, v16bf b, v8f c) {
  // D = A(16x32 bf16) * B(32x16 bf16) + C(16x16 f32)
  return __builtin_amdgcn_wmma_f32_16x16x32_bf16(false, a, false, b, (short)0, c, false, false);
}

// ---- kernel 0: f32 -> bf16 convert ----
__global__ __launch_bounds__(256) void cvt_kernel(const float* __restrict__ src,
                                                  unsigned short* __restrict__ dst, int n) {
  int i = blockIdx.x * blockDim.x + threadIdx.x;
  int stride = gridDim.x * blockDim.x;
  for (; i < n; i += stride) dst[i] = f2bf(src[i]);
}

// =====================================================================
// GEMM core: block = 256 threads (8 waves); block tile = 64 rows x 128 cols.
// Each wave owns a 64x16 column strip: 4 WMMAs per 32-wide k-step, one
// B fragment (reused 4x), A fragments from a cooperatively staged LDS tile.
// =====================================================================

// ---- kernel 1: QKV GEMM  out[s, m, o] = sum_k x[s,m,k] * Wqkv[o,k] ----
__global__ __launch_bounds__(256) void qkv_gemm(const unsigned short* __restrict__ xb,
                                                const unsigned short* __restrict__ wq,
                                                unsigned short* __restrict__ qkvraw) {
  __shared__ __align__(16) unsigned short atile[64 * 32];   // 4KB A tile, shared by 8 waves
  const int lane = threadIdx.x & 31;
  const int wave = threadIdx.x >> 5;
  const int nl = lane & 15;
  const int hh = lane >> 4;
  const int s = blockIdx.z;
  const int mbase = blockIdx.y * 64;
  const int col = (blockIdx.x * 8 + wave) * 16 + nl;

  const unsigned short* arow = xb + ((size_t)s * M_QKV + mbase) * K_DIM;
  const unsigned short* wcol = wq + (size_t)col * K_DIM;
  const int t = threadIdx.x;
  const int srow = t >> 2, sck = (t & 3) * 8;   // staging: one 16B copy per thread

  v8f acc[4];
  v8f vzero = {};
  #pragma unroll
  for (int rt = 0; rt < 4; rt++) acc[rt] = vzero;

  for (int kk = 0; kk < K_DIM; kk += 32) {
    STAGE16(atile + t * 8, arow + (size_t)srow * K_DIM + kk + sck);
    STAGE_FENCE();
    __syncthreads();
    // one B fragment, reused for 4 row-tiles
    v16bf bfrag = cat8(*(const v8bf*)(wcol + kk + hh * 8), *(const v8bf*)(wcol + kk + hh * 8 + 16));
    __builtin_prefetch(wcol + kk + 64, 0, 1);
    #pragma unroll
    for (int rt = 0; rt < 4; rt++) {
      v8bf alo = *(const v8bf*)(atile + (rt * 16 + nl) * 32 + hh * 8);
      v8bf ahi = *(const v8bf*)(atile + (rt * 16 + nl) * 32 + hh * 8 + 16);
      acc[rt] = wmma_bf16(cat8(alo, ahi), bfrag, acc[rt]);
    }
    __syncthreads();
  }
  #pragma unroll
  for (int rt = 0; rt < 4; rt++) {
    #pragma unroll
    for (int r = 0; r < 8; r++) {
      int mrow = mbase + rt * 16 + r + 8 * hh;
      qkvraw[((size_t)s * M_QKV + mrow) * N_QKV + col] = f2bf(acc[rt][r]);
    }
  }
}

// ---- kernel 2: per-head LayerNorm, scatter into WMMA-friendly layouts ----
// q,k -> [s][t][B][H][N][HD] ; v -> [s][2][B][H][HD][N]  (pre-transposed for P*V B-fragments)
__global__ __launch_bounds__(256) void ln_kernel(const unsigned short* __restrict__ qkvraw,
                                                 const float* __restrict__ gg,
                                                 const float* __restrict__ bbias,
                                                 unsigned short* __restrict__ qkvn) {
  const int lane = threadIdx.x & 31;
  const int wave = threadIdx.x >> 5;
  long seg = (long)blockIdx.x * 8 + wave;          // 2*4096*36 segments
  int gidx = (int)(seg % 36);
  long tmp = seg / 36;
  int m = (int)(tmp & 4095);
  int s = (int)(tmp >> 12);
  int tq = gidx / 12, h = gidx % 12;

  const unsigned short* src = qkvraw + ((size_t)s * M_QKV + m) * N_QKV + tq * Cdim + h * HDd;
  float x0 = bf2f(src[lane]);
  float x1 = bf2f(src[lane + 32]);
  float sum = x0 + x1;
  #pragma unroll
  for (int off = 1; off < 32; off <<= 1) sum += __shfl_xor(sum, off, 32);
  float mu = sum * (1.0f / 64.0f);
  float sq = x0 * x0 + x1 * x1;
  #pragma unroll
  for (int off = 1; off < 32; off <<= 1) sq += __shfl_xor(sq, off, 32);
  float var = sq * (1.0f / 64.0f) - mu * mu;
  float rstd = rsqrtf(var + 1e-5f);
  float y0 = (x0 - mu) * rstd * gg[lane] + bbias[lane];
  float y1 = (x1 - mu) * rstd * gg[lane + 32] + bbias[lane + 32];

  int b = m >> 10, n = m & 1023;
  size_t d0, d1;
  if (tq < 2) {
    size_t base = ((((size_t)(s * 3 + tq) * Bq + b) * Hh + h) * Nseq + n) * HDd;
    d0 = base + lane; d1 = base + lane + 32;
  } else {
    size_t base = ((((size_t)(s * 3 + 2) * Bq + b) * Hh + h) * HDd) * Nseq + n;
    d0 = base + (size_t)lane * Nseq; d1 = base + (size_t)(lane + 32) * Nseq;
  }
  qkvn[d0] = f2bf(y0);
  qkvn[d1] = f2bf(y1);
}

// ---- kernel 3: flash attention ----
// Block = 8 waves handling 8 consecutive Q tiles of one (o,b,h);
// K (32x64) and V^T (64x32) tiles staged cooperatively in LDS each step.
__global__ __launch_bounds__(256) void attn_kernel(const unsigned short* __restrict__ qkvn,
                                                   unsigned short* __restrict__ ctxb) {
  __shared__ __align__(16) unsigned short kbuf[32 * 64];      // [key][hd]   4KB
  __shared__ __align__(16) unsigned short vbuf[64 * 32];      // [hd][key]   4KB
  __shared__ __align__(16) unsigned short pbuf[8 * 16 * 32];  // per-wave P  8KB
  const int lane = threadIdx.x & 31;
  const int wave = threadIdx.x >> 5;
  const int nl = lane & 15;
  const int hh = lane >> 4;
  const int t = threadIdx.x;

  int tmp = blockIdx.x >> 3;                   // 96 (o,b,h) groups
  int qt = (blockIdx.x & 7) * 8 + wave;        // this wave's Q tile (0..63)
  int h = tmp % 12;
  int bo = tmp / 12;
  int b = bo & 3;
  int o = bo >> 2;                             // 0 -> context_b, 1 -> context_a
  int sq = 1 - o, skv = o;

  const unsigned short* qptr = qkvn + ((((size_t)(sq * 3 + 0) * Bq + b) * Hh + h) * Nseq + qt * 16) * HDd;
  const unsigned short* kptr = qkvn + (((size_t)(skv * 3 + 1) * Bq + b) * Hh + h) * (size_t)Nseq * HDd;
  const unsigned short* vtp  = qkvn + (((size_t)(skv * 3 + 2) * Bq + b) * Hh + h) * (size_t)HDd * Nseq;
  unsigned short* myl = pbuf + wave * 512;

  // Q A-fragments (hd 0..31 and 32..63), held in registers all loop long
  const unsigned short* qrow = qptr + (size_t)nl * HDd;
  v16bf aq0 = cat8(*(const v8bf*)(qrow + hh * 8),      *(const v8bf*)(qrow + hh * 8 + 16));
  v16bf aq1 = cat8(*(const v8bf*)(qrow + 32 + hh * 8), *(const v8bf*)(qrow + 32 + hh * 8 + 16));

  // staging offsets (one 16B copy per thread per tile)
  const int krow = t >> 3, koff = (t & 7) * 8;   // K: 32 rows x 64 hd
  const int vrow = t >> 2, voff = (t & 3) * 8;   // V^T: 64 rows x 32 keys

  float mi[8], li[8];
  v8f acc[4];
  v8f vzero = {};
  #pragma unroll
  for (int r = 0; r < 8; r++) { mi[r] = -3.0e38f; li[r] = 0.0f; }
  #pragma unroll
  for (int j = 0; j < 4; j++) acc[j] = vzero;

  const float scale = 0.125f;   // 64^-0.5

  for (int kt = 0; kt < Nseq; kt += 32) {
    // cooperative async staging of K and V^T tiles
    STAGE16(kbuf + t * 8, kptr + (size_t)(kt + krow) * HDd + koff);
    STAGE16(vbuf + t * 8, vtp + (size_t)vrow * Nseq + kt + voff);
    STAGE_FENCE();
    __syncthreads();

    // S = Q*K^T from LDS fragments
    v16bf bk0a = cat8(*(const v8bf*)(kbuf + nl * 64 + hh * 8), *(const v8bf*)(kbuf + nl * 64 + hh * 8 + 16));
    v16bf bk1a = cat8(*(const v8bf*)(kbuf + nl * 64 + 32 + hh * 8), *(const v8bf*)(kbuf + nl * 64 + 32 + hh * 8 + 16));
    v16bf bk0b = cat8(*(const v8bf*)(kbuf + (16 + nl) * 64 + hh * 8), *(const v8bf*)(kbuf + (16 + nl) * 64 + hh * 8 + 16));
    v16bf bk1b = cat8(*(const v8bf*)(kbuf + (16 + nl) * 64 + 32 + hh * 8), *(const v8bf*)(kbuf + (16 + nl) * 64 + 32 + hh * 8 + 16));

    v8f c0 = vzero, c1 = vzero;
    c0 = wmma_bf16(aq0, bk0a, c0);
    c0 = wmma_bf16(aq1, bk1a, c0);
    c1 = wmma_bf16(aq0, bk0b, c1);
    c1 = wmma_bf16(aq1, bk1b, c1);

    // online softmax: rows live on half-waves (row = r + 8*hh)
    #pragma unroll
    for (int r = 0; r < 8; r++) {
      float e0 = c0[r] * scale, e1 = c1[r] * scale;
      float mx = fmaxf(e0, e1);
      #pragma unroll
      for (int off = 1; off < 16; off <<= 1) mx = fmaxf(mx, __shfl_xor(mx, off, 32));
      float nm = fmaxf(mi[r], mx);
      float al = __expf(mi[r] - nm);
      float p0 = __expf(e0 - nm);
      float p1 = __expf(e1 - nm);
      float rs = p0 + p1;
      #pragma unroll
      for (int off = 1; off < 16; off <<= 1) rs += __shfl_xor(rs, off, 32);
      li[r] = li[r] * al + rs;
      mi[r] = nm;
      acc[0][r] *= al; acc[1][r] *= al; acc[2][r] *= al; acc[3][r] *= al;
      int row = r + 8 * hh;
      myl[row * 32 + nl]      = f2bf(p0);
      myl[row * 32 + nl + 16] = f2bf(p1);
    }

    // reload P as A-fragment (lane = row, K = key offset)
    v8bf plo = *(const v8bf*)(myl + nl * 32 + hh * 8);
    v8bf phi = *(const v8bf*)(myl + nl * 32 + hh * 8 + 16);
    v16bf pa = cat8(plo, phi);

    // ctx += P * V from LDS V^T tile
    #pragma unroll
    for (int j = 0; j < 4; j++) {
      v16bf vb = cat8(*(const v8bf*)(vbuf + (j * 16 + nl) * 32 + hh * 8), *(const v8bf*)(vbuf + (j * 16 + nl) * 32 + hh * 8 + 16));
      acc[j] = wmma_bf16(pa, vb, acc[j]);
    }
    __syncthreads();
  }

  // normalize and store ctx as bf16 [o][B][N][C] with C index = h*64 + d
  #pragma unroll
  for (int j = 0; j < 4; j++) {
    #pragma unroll
    for (int r = 0; r < 8; r++) {
      float v = acc[j][r] / li[r];
      int row = qt * 16 + r + 8 * hh;
      size_t idx = (((size_t)o * Bq + b) * Nseq + row) * Cdim + h * HDd + j * 16 + nl;
      ctxb[idx] = f2bf(v);
    }
  }
}

// ---- kernel 4: output projection + bias + faithful (non-transposed) q residual ----
__global__ __launch_bounds__(256) void proj_gemm(const unsigned short* __restrict__ ctxb,
                                                 const unsigned short* __restrict__ wp,
                                                 const float* __restrict__ bproj,
                                                 const unsigned short* __restrict__ qkvn,
                                                 float* __restrict__ out) {
  __shared__ __align__(16) unsigned short atile[64 * 32];
  const int lane = threadIdx.x & 31;
  const int wave = threadIdx.x >> 5;
  const int nl = lane & 15;
  const int hh = lane >> 4;
  const int mbase = blockIdx.y * 64;                 // M = 2*B*N = 8192
  const int col = (blockIdx.x * 8 + wave) * 16 + nl; // 768 cols

  const unsigned short* arow = ctxb + (size_t)mbase * K_DIM;
  const unsigned short* wcol = wp + (size_t)col * K_DIM;
  const int t = threadIdx.x;
  const int srow = t >> 2, sck = (t & 3) * 8;

  v8f acc[4];
  v8f vzero = {};
  #pragma unroll
  for (int rt = 0; rt < 4; rt++) acc[rt] = vzero;

  for (int kk = 0; kk < K_DIM; kk += 32) {
    STAGE16(atile + t * 8, arow + (size_t)srow * K_DIM + kk + sck);
    STAGE_FENCE();
    __syncthreads();
    v16bf bfrag = cat8(*(const v8bf*)(wcol + kk + hh * 8), *(const v8bf*)(wcol + kk + hh * 8 + 16));
    __builtin_prefetch(wcol + kk + 64, 0, 1);
    #pragma unroll
    for (int rt = 0; rt < 4; rt++) {
      v8bf alo = *(const v8bf*)(atile + (rt * 16 + nl) * 32 + hh * 8);
      v8bf ahi = *(const v8bf*)(atile + (rt * 16 + nl) * 32 + hh * 8 + 16);
      acc[rt] = wmma_bf16(cat8(alo, ahi), bfrag, acc[rt]);
    }
    __syncthreads();
  }

  #pragma unroll
  for (int rt = 0; rt < 4; rt++) {
    #pragma unroll
    for (int r = 0; r < 8; r++) {
      int m = mbase + rt * 16 + r + 8 * hh;
      int o = m >> 12;             // which output tensor
      int bb = (m >> 10) & 3;
      int n = m & 1023;
      int flat = n * Cdim + col;   // raw flat index within [N,C]
      int h = flat >> 16;          // / (N*HD) = 65536
      int rem = flat & 65535;
      int n2 = rem >> 6;
      int d = rem & 63;
      int sres = 1 - o;
      float qv = bf2f(qkvn[((((size_t)(sres * 3 + 0) * Bq + bb) * Hh + h) * Nseq + n2) * HDd + d]);
      out[(size_t)m * Cdim + col] = acc[rt][r] + bproj[col] + qv;
    }
  }
}

extern "C" void kernel_launch(void* const* d_in, const int* in_sizes, int n_in,
                              void* d_out, int out_size, void* d_ws, size_t ws_size,
                              hipStream_t stream) {
  (void)in_sizes; (void)n_in; (void)out_size; (void)ws_size;
  const float* before = (const float*)d_in[0];
  const float* after  = (const float*)d_in[1];
  const float* Wqkv   = (const float*)d_in[2];
  const float* ln_g   = (const float*)d_in[3];
  const float* ln_b   = (const float*)d_in[4];
  const float* Wproj  = (const float*)d_in[5];
  const float* bproj  = (const float*)d_in[6];

  // workspace layout (bf16 stored as ushort)
  unsigned short* xb     = (unsigned short*)d_ws;                       // [2][4096][768]
  unsigned short* wq     = xb + (size_t)2 * M_QKV * K_DIM;              // [2304][768]
  unsigned short* wp     = wq + (size_t)N_QKV * K_DIM;                  // [768][768]
  unsigned short* qkvraw = wp + (size_t)K_DIM * K_DIM;                  // [2][4096][2304]
  unsigned short* qkvn   = qkvraw + (size_t)2 * M_QKV * N_QKV;          // [2][3][B][H][...] (v transposed)
  unsigned short* ctxb   = qkvn + (size_t)2 * M_QKV * N_QKV;            // [2][B][N][C]

  const int NX = M_QKV * K_DIM;           // 3,145,728 per input
  cvt_kernel<<<1536, 256, 0, stream>>>(before, xb, NX);
  cvt_kernel<<<1536, 256, 0, stream>>>(after, xb + (size_t)NX, NX);
  cvt_kernel<<<864, 256, 0, stream>>>(Wqkv, wq, N_QKV * K_DIM);
  cvt_kernel<<<288, 256, 0, stream>>>(Wproj, wp, K_DIM * K_DIM);

  qkv_gemm<<<dim3(N_QKV / 128, M_QKV / 64, 2), 256, 0, stream>>>(xb, wq, qkvraw);
  ln_kernel<<<(2 * M_QKV * 36) / 8, 256, 0, stream>>>(qkvraw, ln_g, ln_b, qkvn);
  attn_kernel<<<(2 * Bq * Hh * (Nseq / 16)) / 8, 256, 0, stream>>>(qkvn, ctxb);
  proj_gemm<<<dim3(K_DIM / 128, (2 * M_QKV) / 64), 256, 0, stream>>>(ctxb, wp, bproj, qkvn, (float*)d_out);
}
